// ScaledDotProductAttention_86672440033785
// MI455X (gfx1250) — matrix operand order repaired and checked by reading, not compiled, not run on hardware
//
#include <hip/hip_runtime.h>

typedef __attribute__((ext_vector_type(16))) _Float16 v16h;
typedef __attribute__((ext_vector_type(8)))  _Float16 v8h;
typedef __attribute__((ext_vector_type(4)))  _Float16 v4h;
typedef __attribute__((ext_vector_type(8)))  float    v8f;
typedef __attribute__((ext_vector_type(4)))  float    f32x4;

#define SDIM 2048
#define DDIM 64
#define QBLK 128   // q rows per workgroup (8 waves x 16)
#define KSTR 72    // K LDS row stride in f16 (144B: 16B-aligned, conflict-free)
#define VSTR 40    // V^T LDS row stride in f16 (80B)
#define PSTR 40    // P LDS row stride in f16 (80B)
#define NEGM (-1.0e30f)

// ds_swizzle_b32 xor-mode: new_lane = ((lane & 0x1F) | 0) ^ M  (M<16 stays in half)
template <int M>
__device__ __forceinline__ float swz_xor(float v) {
    constexpr int imm = (M << 10) | 0x1F;
    return __uint_as_float(
        (unsigned)__builtin_amdgcn_ds_swizzle((int)__float_as_uint(v), imm));
}

__device__ __forceinline__ float redmax16(float v) {
    v = fmaxf(v, swz_xor<8>(v));
    v = fmaxf(v, swz_xor<4>(v));
    v = fmaxf(v, swz_xor<2>(v));
    v = fmaxf(v, swz_xor<1>(v));
    return v;
}

__device__ __forceinline__ v8f zero8() {
    v8f z;
#pragma unroll
    for (int i = 0; i < 8; ++i) z[i] = 0.0f;
    return z;
}

__launch_bounds__(256)
__global__ void fa_fwd_causal_kernel(const float* __restrict__ Q,
                                     const float* __restrict__ K,
                                     const float* __restrict__ V,
                                     float* __restrict__ O)
{
    __shared__ __align__(16) _Float16 Kl[32 * KSTR];      // K tile, [key][d]
    __shared__ __align__(16) _Float16 Vl[64 * VSTR];      // V tile transposed, [d][key]
    __shared__ __align__(16) _Float16 Pl[8 * 16 * PSTR];  // per-wave P scratch, [m][k]

    const int tid  = threadIdx.x;
    const int wave = tid >> 5;
    const int lane = tid & 31;
    const int n    = lane & 15;   // column index within 16-wide tiles
    const int hi   = lane >> 4;   // lane half

    const int qb = blockIdx.x * QBLK;    // workgroup q block base
    const int bh = blockIdx.y;           // fused batch*head
    const int qt = qb + wave * 16;       // this wave's 16-row q tile base

    const size_t head = (size_t)bh * SDIM * DDIM;
    const float* Qg = Q + head;
    const float* Kg = K + head;
    const float* Vg = V + head;
    float*       Og = O + head;

    const float scale = 0.125f;  // 1/sqrt(64)

    // ---- Q tile as two 16x32 f16 A-fragments (D = 64 = 2 k-steps) ----
    // A layout: lane row m = lane&15; hi=0 holds K {k0..k0+7, k0+16..k0+23}, k0=0;
    //           hi=1 the same with k0=8 (16-bit A-matrix table).
    v16h aq[2];
    {
        const float* qrow = Qg + (size_t)(qt + n) * DDIM;
#pragma unroll
        for (int ks = 0; ks < 2; ++ks) {
            const int d0 = ks * 32 + hi * 8;
#pragma unroll
            for (int j = 0; j < 8; ++j)
                aq[ks][j]     = (_Float16)(qrow[d0 + j] * scale);
#pragma unroll
            for (int j = 0; j < 8; ++j)
                aq[ks][8 + j] = (_Float16)(qrow[d0 + 16 + j] * scale);
        }
    }

    // constant all-ones B fragment: rowsum(P) = P x ones via the matrix unit
    v16h bones;
#pragma unroll
    for (int j = 0; j < 16; ++j) bones[j] = (_Float16)1.0f;

    // ---- flash-attention running state ----
    v8f o[4];                      // O accumulators: [d-tile][row], C-layout
#pragma unroll
    for (int dt = 0; dt < 4; ++dt) o[dt] = zero8();
    float mrun[8], lrun[8];
#pragma unroll
    for (int r = 0; r < 8; ++r) { mrun[r] = NEGM; lrun[r] = 0.0f; }

    const int kbmax = (qb + QBLK - 1) >> 5;   // last key block index (causal bound)
    for (int kb = 0; kb <= kbmax; ++kb) {
        const int kbase = kb * 32;

        __syncthreads();   // previous compute done before LDS overwrite

        // ---- cooperative stage: K tile (row-major) + V tile (transposed), f32->f16 ----
#pragma unroll
        for (int i = 0; i < 2; ++i) {
            const int f4  = tid + i * 256;       // 0..511 float4 chunks of the 32x64 tile
            const int row = f4 >> 4;             // key within block
            const int dd  = (f4 & 15) * 4;       // d offset
            const size_t goff = (size_t)(kbase + row) * DDIM + dd;
            f32x4 kv = *(const f32x4*)(Kg + goff);
            v4h kh;
            kh[0] = (_Float16)kv.x; kh[1] = (_Float16)kv.y;
            kh[2] = (_Float16)kv.z; kh[3] = (_Float16)kv.w;
            *(v4h*)&Kl[row * KSTR + dd] = kh;
            f32x4 vv = *(const f32x4*)(Vg + goff);
            Vl[(dd + 0) * VSTR + row] = (_Float16)vv.x;
            Vl[(dd + 1) * VSTR + row] = (_Float16)vv.y;
            Vl[(dd + 2) * VSTR + row] = (_Float16)vv.z;
            Vl[(dd + 3) * VSTR + row] = (_Float16)vv.w;
        }
        if (kb < kbmax) {  // speculative prefetch of next K/V tile (stays in-bounds)
            const size_t noff = (size_t)(kbase + 32) * DDIM + tid * 8;
            __builtin_prefetch(Kg + noff, 0, 1);
            __builtin_prefetch(Vg + noff, 0, 1);
        }
        __syncthreads();

        // wave-uniform causal skip: keep EXEC all-ones around WMMA
        if (kbase > qt + 15) continue;

        // ---- S = Q * K^T  (16x32 scores = two 16x16 C tiles) ----
        v8f c[2];
#pragma unroll
        for (int nt = 0; nt < 2; ++nt) {
            c[nt] = zero8();
#pragma unroll
            for (int ks = 0; ks < 2; ++ks) {
                // B layout: lane col = lane&15 (key row nt*16+n), hi holds K {8hi..8hi+7, 16+8hi..16+8hi+7}
                const int base = (nt * 16 + n) * KSTR + ks * 32 + hi * 8;
                v8h b0 = *(const v8h*)&Kl[base];
                v8h b1 = *(const v8h*)&Kl[base + 16];
                v16h bk;
#pragma unroll
                for (int j = 0; j < 8; ++j) { bk[j] = b0[j]; bk[8 + j] = b1[j]; }
                c[nt] = __builtin_amdgcn_wmma_f32_16x16x32_f16(
                    false, aq[ks], false, bk, (short)0, c[nt], false, false);
            }
        }

        // ---- per-row max (mask only on diagonal blocks) ----
        const bool full = (kbase + 31 <= qt);   // wave-uniform: no mask needed
        float mt[8];
        if (full) {
#pragma unroll
            for (int r = 0; r < 8; ++r) mt[r] = fmaxf(c[0][r], c[1][r]);
        } else {
            const int q0 = qt + hi * 8;          // q row for vgpr index r
#pragma unroll
            for (int r = 0; r < 8; ++r) mt[r] = NEGM;
#pragma unroll
            for (int nt = 0; nt < 2; ++nt) {
                const int kcol = kbase + nt * 16 + n;
#pragma unroll
                for (int r = 0; r < 8; ++r) {
                    float v = (kcol > q0 + r) ? NEGM : c[nt][r];
                    c[nt][r] = v;
                    mt[r] = fmaxf(mt[r], v);
                }
            }
        }
#pragma unroll
        for (int r = 0; r < 8; ++r) mt[r] = redmax16(mt[r]);  // ds_swizzle butterflies

        float alpha[8];
#pragma unroll
        for (int r = 0; r < 8; ++r) {
            const float mn = fmaxf(mrun[r], mt[r]);
            alpha[r] = __expf(mrun[r] - mn);
            mrun[r]  = mn;
        }
        // P = exp(S - m)
#pragma unroll
        for (int r = 0; r < 8; ++r) {
            c[0][r] = __expf(c[0][r] - mrun[r]);
            c[1][r] = __expf(c[1][r] - mrun[r]);
        }
        // rescale running O
#pragma unroll
        for (int dt = 0; dt < 4; ++dt)
#pragma unroll
            for (int r = 0; r < 8; ++r)
                o[dt][r] *= alpha[r];

        // ---- re-layout P: C-layout regs -> LDS row-major -> A-fragment ----
        _Float16* pw = &Pl[wave * 16 * PSTR];
#pragma unroll
        for (int nt = 0; nt < 2; ++nt)
#pragma unroll
            for (int r = 0; r < 8; ++r)
                pw[(r + 8 * hi) * PSTR + nt * 16 + n] = (_Float16)c[nt][r];

        v16h ap;
        {
            const int base = n * PSTR + hi * 8;
            v8h p0 = *(const v8h*)&pw[base];
            v8h p1 = *(const v8h*)&pw[base + 16];
#pragma unroll
            for (int j = 0; j < 8; ++j) { ap[j] = p0[j]; ap[8 + j] = p1[j]; }
        }

        // ---- row sums on the matrix unit: csum = P x ones (replicated per lane) ----
        v8f csum = __builtin_amdgcn_wmma_f32_16x16x32_f16(
            false, ap, false, bones, (short)0, zero8(), false, false);
#pragma unroll
        for (int r = 0; r < 8; ++r)
            lrun[r] = lrun[r] * alpha[r] + csum[r];

        // ---- O += P * V  (V^T staged in LDS: contiguous keys per d-row) ----
#pragma unroll
        for (int dt = 0; dt < 4; ++dt) {
            const int base = (dt * 16 + n) * VSTR + hi * 8;
            v8h b0 = *(const v8h*)&Vl[base];
            v8h b1 = *(const v8h*)&Vl[base + 16];
            v16h bv;
#pragma unroll
            for (int j = 0; j < 8; ++j) { bv[j] = b0[j]; bv[8 + j] = b1[j]; }
            o[dt] = __builtin_amdgcn_wmma_f32_16x16x32_f16(
                false, ap, false, bv, (short)0, o[dt], false, false);
        }
    }

    // ---- normalize and store O (fp32) ----
#pragma unroll
    for (int r = 0; r < 8; ++r) {
        const float inv = 1.0f / lrun[r];   // diagonal always unmasked => lrun > 0
#pragma unroll
        for (int dt = 0; dt < 4; ++dt)
            Og[(size_t)(qt + r + 8 * hi) * DDIM + dt * 16 + n] = o[dt][r] * inv;
    }
}

extern "C" void kernel_launch(void* const* d_in, const int* in_sizes, int n_in,
                              void* d_out, int out_size, void* d_ws, size_t ws_size,
                              hipStream_t stream) {
    (void)in_sizes; (void)n_in; (void)out_size; (void)d_ws; (void)ws_size;
    const float* Q = (const float*)d_in[0];
    const float* K = (const float*)d_in[1];
    const float* V = (const float*)d_in[2];
    float* O = (float*)d_out;

    dim3 grid(SDIM / QBLK, 4 * 16);   // (q blocks, B*H)
    dim3 block(256);                  // 8 waves of 32
    fa_fwd_causal_kernel<<<grid, block, 0, stream>>>(Q, K, V, O);
}
